// minLSTM_72945724555722
// MI455X (gfx1250) — matrix operand order repaired and checked by reading, compiled, not run on hardware
//
#include <hip/hip_runtime.h>
#include <math.h>

// ---------------------------------------------------------------------------
// minLSTM forward for MI455X (gfx1250, wave32, WMMA 16x16x32 bf16)
//   B=4, S=4096, D=1024, Din=2048
//   k1: fused GEMM1 (x@W1+b1 for h/f/i) + gate math -> log_coeffs/log_values
//   k2a/k2b/k2c: chunked log-space scan over S (32 chunks of 128)
//   k3: GEMM2 exp(log_h)@W2 + b2 -> out, A staged via async-to-LDS (gfx1250)
// ---------------------------------------------------------------------------

typedef __bf16 v8bf  __attribute__((ext_vector_type(8)));
typedef __bf16 v16bf __attribute__((ext_vector_type(16)));
typedef float  v8f   __attribute__((ext_vector_type(8)));

#define S_LEN   4096
#define B_SZ    4
#define D_MODEL 1024
#define D_IN    2048
#define NROWS   (B_SZ * S_LEN)   // 16384
#define NCHUNK  32
#define CHUNK   128              // S_LEN / NCHUNK

// float -> bf16 bits, round-to-nearest-even
__device__ __forceinline__ unsigned short f2bf(float f) {
  union { float f; unsigned u; } v; v.f = f;
  unsigned r = v.u + 0x7FFFu + ((v.u >> 16) & 1u);
  return (unsigned short)(r >> 16);
}

// build a 16-element bf16 fragment from two contiguous 8-halfword LDS runs
__device__ __forceinline__ v16bf frag16(const unsigned short* p0,
                                        const unsigned short* p1) {
  v8bf lo = *reinterpret_cast<const v8bf*>(p0);
  v8bf hi = *reinterpret_cast<const v8bf*>(p1);
  v16bf r;
#pragma unroll
  for (int e = 0; e < 8; ++e) { r[e] = lo[e]; r[e + 8] = hi[e]; }
  return r;
}

__device__ __forceinline__ float softplusf(float z) {
  return fmaxf(z, 0.f) + log1pf(__expf(-fabsf(z)));
}
__device__ __forceinline__ float loggf(float x) {  // log_g of reference
  return (x >= 0.f) ? __logf(x + 0.5f) : -softplusf(-x);
}
__device__ __forceinline__ float logaddexpf_(float u, float w) {
  float m = fmaxf(u, w);
  if (m == -INFINITY) return -INFINITY;
  return m + log1pf(__expf(fminf(u, w) - m));
}

// gfx1250 async copy: 16 bytes global -> LDS, tracked by ASYNCcnt.
// ldsoff: LDS byte offset (low 32 bits of a flat LDS pointer per ISA 10.2).
__device__ __forceinline__ void async_g2l_b128(unsigned ldsoff,
                                               unsigned long long gaddr) {
  asm volatile("global_load_async_to_lds_b128 %0, %1, off"
               :: "v"(ldsoff), "v"(gaddr) : "memory");
}
__device__ __forceinline__ void wait_asynccnt0() {
  asm volatile("s_wait_asynccnt 0x0" ::: "memory");
}

// ---------------------------------------------------------------------------
// Kernel 1: fused proj GEMM + gate math.
// Block: 256 thr (8 wave32). Tile: M=128 rows, N=32 cols (per gate), K=32.
// Wave w owns rows [w*16, w*16+16); accumulates 3 gates x 2 n-tiles.
// ---------------------------------------------------------------------------
__global__ __launch_bounds__(256) void k_gemm1_gates(
    const float* __restrict__ x, const float* __restrict__ W1,
    const float* __restrict__ b1, float* __restrict__ logc,
    float* __restrict__ logv) {
  __shared__ unsigned short lA[128][40];      // x tile, bf16, padded rows
  __shared__ unsigned short lB[3][32][40];    // W1 tiles transposed [n][k]

  const int t    = threadIdx.x;
  const int wv   = t >> 5;
  const int lane = t & 31;
  const int hf   = lane >> 4;       // half-wave select
  const int l15  = lane & 15;
  const int M0   = blockIdx.x * 128;
  const int N0   = blockIdx.y * 32;

  v8f acc[3][2];
#pragma unroll
  for (int g = 0; g < 3; ++g)
#pragma unroll
    for (int nt = 0; nt < 2; ++nt)
#pragma unroll
      for (int e = 0; e < 8; ++e) acc[g][nt][e] = 0.f;

  for (int k0 = 0; k0 < D_MODEL; k0 += 32) {
    __syncthreads();
    // stage A: 128x32 f32 -> bf16 (coalesced float4 reads)
#pragma unroll
    for (int i = 0; i < 4; ++i) {
      int q = t + i * 256;                 // 1024 float4 chunks
      int m = q >> 3, kq = (q & 7) * 4;
      const float4 v = *reinterpret_cast<const float4*>(
          x + (size_t)(M0 + m) * D_MODEL + k0 + kq);
      unsigned short* dst = &lA[m][kq];
      dst[0] = f2bf(v.x); dst[1] = f2bf(v.y);
      dst[2] = f2bf(v.z); dst[3] = f2bf(v.w);
    }
    // stage B: 3 gates x 32k x 32n, transposed into [n][k]
#pragma unroll
    for (int i = 0; i < 3; ++i) {
      int q = t + i * 256;                 // 768 float4 chunks
      int g = q >> 8, rem = q & 255;
      int k = rem >> 3, nq = (rem & 7) * 4;
      const float4 v = *reinterpret_cast<const float4*>(
          W1 + (size_t)(k0 + k) * (3 * D_IN) + g * D_IN + N0 + nq);
      lB[g][nq + 0][k] = f2bf(v.x);
      lB[g][nq + 1][k] = f2bf(v.y);
      lB[g][nq + 2][k] = f2bf(v.z);
      lB[g][nq + 3][k] = f2bf(v.w);
    }
    __syncthreads();

    if (k0 + 32 < D_MODEL)  // global_prefetch_b8 of next A tile
      __builtin_prefetch(x + (size_t)(M0 + (t >> 1)) * D_MODEL + k0 + 32, 0, 1);

    // A fragment: row m = l15 (of this wave's strip), K runs {8*hf, 16+8*hf}
    const v16bf a = frag16(&lA[wv * 16 + l15][8 * hf],
                           &lA[wv * 16 + l15][16 + 8 * hf]);
#pragma unroll
    for (int g = 0; g < 3; ++g) {
#pragma unroll
      for (int nt = 0; nt < 2; ++nt) {
        // B fragment: col n = l15, K runs {8*hf, 16+8*hf}
        const v16bf b = frag16(&lB[g][nt * 16 + l15][8 * hf],
                               &lB[g][nt * 16 + l15][16 + 8 * hf]);
        acc[g][nt] = __builtin_amdgcn_wmma_f32_16x16x32_bf16(
            false, a, false, b, (short)0, acc[g][nt], false, false);
      }
    }
  }

  // epilogue: bias + gate algebra, write log_coeffs / log_values
  const int mbase = M0 + wv * 16 + 8 * hf;
#pragma unroll
  for (int nt = 0; nt < 2; ++nt) {
    const int n = N0 + nt * 16 + l15;
    const float bh = b1[n], bfg = b1[D_IN + n], big = b1[2 * D_IN + n];
#pragma unroll
    for (int r = 0; r < 8; ++r) {
      const int row = mbase + r;
      float h = acc[0][nt][r] + bh;
      float f = acc[1][nt][r] + bfg;
      float i = acc[2][nt][r] + big;
      float diff = softplusf(-f) - softplusf(-i);
      float lc = -softplusf(diff);
      float lv = -softplusf(-diff) + loggf(h);
      size_t o = (size_t)row * D_IN + n;
      logc[o] = lc;
      logv[o] = lv;
    }
  }
}

// ---------------------------------------------------------------------------
// Scan pass 1: per (b, chunk, channel) compute chunk transform (A, V).
// ---------------------------------------------------------------------------
__global__ __launch_bounds__(256) void k_scan1(
    const float* __restrict__ logc, const float* __restrict__ logv,
    float* __restrict__ cA, float* __restrict__ cV) {
  int id = blockIdx.x * 256 + threadIdx.x;   // B*NCHUNK*D_IN = 262144
  int c  = id & (D_IN - 1);
  int ch = (id >> 11) & (NCHUNK - 1);
  int b  = id >> 16;
  float a = 0.f, v = -INFINITY;
  size_t base = ((size_t)b * S_LEN + ch * CHUNK) * D_IN + c;
  for (int s = 0; s < CHUNK; ++s) {
    float lc = logc[base], lv = logv[base];
    a += lc;
    v = logaddexpf_(v + lc, lv);
    base += D_IN;
  }
  int o = (b * NCHUNK + ch) * D_IN + c;
  cA[o] = a;
  cV[o] = v;
}

// ---------------------------------------------------------------------------
// Scan pass 2: thread per (b, channel): chain chunk carries from log_g(init_h)
// storing the incoming log_h for each chunk.
// ---------------------------------------------------------------------------
__global__ __launch_bounds__(256) void k_scan2(
    const float* __restrict__ cA, const float* __restrict__ cV,
    const float* __restrict__ init_h, float* __restrict__ pref) {
  int id = blockIdx.x * 256 + threadIdx.x;   // B*D_IN = 8192
  int c = id & (D_IN - 1);
  int b = id >> 11;
  float h = loggf(init_h[c]);
  for (int ch = 0; ch < NCHUNK; ++ch) {
    int o = (b * NCHUNK + ch) * D_IN + c;
    pref[o] = h;
    h = logaddexpf_(cA[o] + h, cV[o]);
  }
}

// ---------------------------------------------------------------------------
// Scan pass 3: replay each chunk from its prefix, emit exp(log_h) as bf16
// (GEMM2's A operand) plus next_hidden / next_log_hidden.
// ---------------------------------------------------------------------------
__global__ __launch_bounds__(256) void k_scan3(
    const float* __restrict__ logc, const float* __restrict__ logv,
    const float* __restrict__ pref, unsigned short* __restrict__ hexp,
    float* __restrict__ next_h, float* __restrict__ next_logh) {
  int id = blockIdx.x * 256 + threadIdx.x;
  int c  = id & (D_IN - 1);
  int ch = (id >> 11) & (NCHUNK - 1);
  int b  = id >> 16;
  float h = pref[(b * NCHUNK + ch) * D_IN + c];
  size_t base = ((size_t)b * S_LEN + ch * CHUNK) * D_IN + c;
  for (int s = 0; s < CHUNK; ++s) {
    float lc = logc[base], lv = logv[base];
    h = logaddexpf_(lc + h, lv);
    hexp[base] = f2bf(__expf(h));
    base += D_IN;
  }
  if (ch == NCHUNK - 1) {
    next_h[b * D_IN + c]    = __expf(h);
    next_logh[b * D_IN + c] = h;
  }
}

// ---------------------------------------------------------------------------
// Kernel 3: out = out_h @ W2 + b2.  Same tiling as GEMM1; A operand (bf16)
// is staged with gfx1250 GLOBAL_LOAD_ASYNC_TO_LDS_B128 (ASYNCcnt-tracked).
// ---------------------------------------------------------------------------
__global__ __launch_bounds__(256) void k_gemm2(
    const unsigned short* __restrict__ hexp, const float* __restrict__ W2,
    const float* __restrict__ b2, float* __restrict__ out) {
  __shared__ unsigned short lA[128][40];
  __shared__ unsigned short lB[32][40];

  const int t    = threadIdx.x;
  const int wv   = t >> 5;
  const int lane = t & 31;
  const int hf   = lane >> 4;
  const int l15  = lane & 15;
  const int M0   = blockIdx.x * 128;
  const int N0   = blockIdx.y * 32;

  v8f acc[2];
#pragma unroll
  for (int nt = 0; nt < 2; ++nt)
#pragma unroll
    for (int e = 0; e < 8; ++e) acc[nt][e] = 0.f;

  for (int k0 = 0; k0 < D_IN; k0 += 32) {
    __syncthreads();
    // stage A (already bf16): 128x32 halfwords via async copy to LDS.
    // 512 x 16B lane-transfers; LDS rows are padded (80B) so per-lane LDS
    // addresses handle the non-contiguous destination.
#pragma unroll
    for (int i = 0; i < 2; ++i) {
      int q = t + i * 256;                 // 512 chunks of 8 halfwords
      int m = q >> 2, kq = (q & 3) * 8;
      unsigned ldsoff = (unsigned)(uintptr_t)&lA[m][kq];
      unsigned long long g =
          (unsigned long long)(uintptr_t)(hexp + (size_t)(M0 + m) * D_IN + k0 + kq);
      async_g2l_b128(ldsoff, g);
    }
    // stage B: 32k x 32n f32 -> bf16 transposed (needs conversion: VALU path)
    {
      int k = t >> 3, nq = (t & 7) * 4;
      const float4 v = *reinterpret_cast<const float4*>(
          W2 + (size_t)(k0 + k) * D_MODEL + N0 + nq);
      lB[nq + 0][k] = f2bf(v.x);
      lB[nq + 1][k] = f2bf(v.y);
      lB[nq + 2][k] = f2bf(v.z);
      lB[nq + 3][k] = f2bf(v.w);
    }
    wait_asynccnt0();   // each wave drains its own async transfers
    __syncthreads();    // then cross-wave visibility

    if (k0 + 32 < D_IN)
      __builtin_prefetch(hexp + (size_t)(M0 + (t >> 1)) * D_IN + k0 + 32, 0, 1);

    const v16bf a = frag16(&lA[wv * 16 + l15][8 * hf],
                           &lA[wv * 16 + l15][16 + 8 * hf]);
#pragma unroll
    for (int nt = 0; nt < 2; ++nt) {
      const v16bf b = frag16(&lB[nt * 16 + l15][8 * hf],
                             &lB[nt * 16 + l15][16 + 8 * hf]);
      acc[nt] = __builtin_amdgcn_wmma_f32_16x16x32_bf16(
          false, a, false, b, (short)0, acc[nt], false, false);
    }
  }

  const int mbase = M0 + wv * 16 + 8 * hf;
#pragma unroll
  for (int nt = 0; nt < 2; ++nt) {
    const int n = N0 + nt * 16 + l15;
    const float bb = b2[n];
#pragma unroll
    for (int r = 0; r < 8; ++r)
      out[(size_t)(mbase + r) * D_MODEL + n] = acc[nt][r] + bb;
  }
}

// ---------------------------------------------------------------------------
// Launch.  Workspace layout (bytes):
//   [0,128M)        log_coeffs  f32  (16384 x 2048)
//   [128M,256M)     log_values  f32
//   [256M,320M)     out_h       bf16 (16384 x 2048)
//   [320M, +1M)     chunk carry A
//   [+1M, +2M)      chunk carry V
//   [+2M, +3M)      chunk prefix log_h
// ---------------------------------------------------------------------------
extern "C" void kernel_launch(void* const* d_in, const int* in_sizes, int n_in,
                              void* d_out, int out_size, void* d_ws,
                              size_t ws_size, hipStream_t stream) {
  (void)in_sizes; (void)n_in; (void)out_size; (void)ws_size;
  const float* x      = (const float*)d_in[0];
  const float* W1     = (const float*)d_in[1];
  const float* b1     = (const float*)d_in[2];
  const float* W2     = (const float*)d_in[3];
  const float* b2     = (const float*)d_in[4];
  const float* init_h = (const float*)d_in[5];
  float* out = (float*)d_out;

  char* ws = (char*)d_ws;
  float*          logc = (float*)(ws);
  float*          logv = (float*)(ws + (size_t)134217728);
  unsigned short* hexp = (unsigned short*)(ws + (size_t)268435456);
  float*          cA   = (float*)(ws + (size_t)335544320);
  float*          cV   = (float*)(ws + (size_t)336592896);
  float*          pref = (float*)(ws + (size_t)337641472);

  float* next_h    = out + (size_t)NROWS * D_MODEL;   // (B,1,Din)
  float* next_logh = next_h + (size_t)B_SZ * D_IN;

  k_gemm1_gates<<<dim3(NROWS / 128, D_IN / 32), 256, 0, stream>>>(
      x, W1, b1, logc, logv);
  k_scan1<<<dim3((B_SZ * NCHUNK * D_IN) / 256), 256, 0, stream>>>(
      logc, logv, cA, cV);
  k_scan2<<<dim3((B_SZ * D_IN) / 256), 256, 0, stream>>>(cA, cV, init_h, pref);
  k_scan3<<<dim3((B_SZ * NCHUNK * D_IN) / 256), 256, 0, stream>>>(
      logc, logv, pref, hexp, next_h, next_logh);
  k_gemm2<<<dim3(NROWS / 128, D_MODEL / 32), 256, 0, stream>>>(
      hexp, W2, b2, out);
}
